// SingleFieldMixerBlock_3633542333068
// MI455X (gfx1250) — matrix operand order repaired and checked by reading, not compiled, not run on hardware
//
#include <hip/hip_runtime.h>

// MI455X / gfx1250. Edge MLP (38.7 GFLOP) dominates compute; E/E_new (256MB)
// sets the ~11us HBM floor. Weights are pre-converted once to transposed f16
// (0.78MB, stays in WGP$/L2) so B-fragments stream straight from cache with no
// per-block transpose/convert; A tiles are float4-staged into double-buffered
// LDS; all dense math is v_wmma_f32_16x16x32_f16 with fp32 accumulation.

typedef __attribute__((ext_vector_type(16))) _Float16 v16h;
typedef __attribute__((ext_vector_type(8)))  _Float16 v8h;
typedef __attribute__((ext_vector_type(4)))  _Float16 v4h;
typedef __attribute__((ext_vector_type(8)))  float    v8f;

#define B_    2
#define N_    8192
#define M_    131072
#define D_    128
#define S_    32
#define NT_   128
#define H_    4
#define NODE_ 160

__device__ __forceinline__ float silu_f(float x) { return x / (1.0f + __expf(-x)); }

// A fragment: 16x32 f16 tile, rows [row0,row0+16), k contiguous with 'stride'.
__device__ __forceinline__ v16h load_a_frag(const _Float16* base, int row0, int stride) {
  int lane = threadIdx.x & 31;
  const _Float16* p = base + (size_t)(row0 + (lane & 15)) * stride + ((lane & 16) ? 16 : 0);
  v8h lo = *(const v8h*)p;
  v8h hi = *(const v8h*)(p + 8);
  v16h r;
#pragma unroll
  for (int i = 0; i < 8; ++i) { r[i] = lo[i]; r[i + 8] = hi[i]; }
  return r;
}

// B fragment from LDS transposed tile baseT[n][k] (k contiguous, 'stride' halves).
__device__ __forceinline__ v16h load_b_frag(const _Float16* baseT, int n0, int stride) {
  int lane = threadIdx.x & 31;
  const _Float16* p = baseT + (size_t)(n0 + (lane & 15)) * stride + ((lane & 16) ? 16 : 0);
  v8h lo = *(const v8h*)p;
  v8h hi = *(const v8h*)(p + 8);
  v16h r;
#pragma unroll
  for (int i = 0; i < 8; ++i) { r[i] = lo[i]; r[i + 8] = hi[i]; }
  return r;
}

// B fragment straight from global pre-transposed f16 weight WT[n][Kd].
__device__ __forceinline__ v16h load_b_gl(const _Float16* WT, int Kd, int n, int k0) {
  int lane = threadIdx.x & 31;
  const _Float16* p = WT + (size_t)(n + (lane & 15)) * Kd + k0 + ((lane & 16) ? 16 : 0);
  v8h lo = *(const v8h*)p;
  v8h hi = *(const v8h*)(p + 8);
  v16h r;
#pragma unroll
  for (int i = 0; i < 8; ++i) { r[i] = lo[i]; r[i + 8] = hi[i]; }
  return r;
}

__device__ __forceinline__ v8f wmma32(v16h a, v16h b, v8f c) {
  return __builtin_amdgcn_wmma_f32_16x16x32_f16(false, a, false, b, (short)0, c, false, false);
}

__device__ __forceinline__ float block_reduce128(float v, float* red, int c) {
  red[c] = v; __syncthreads();
  for (int s = 64; s > 0; s >>= 1) { if (c < s) red[c] += red[c + s]; __syncthreads(); }
  float r = red[0]; __syncthreads();
  return r;
}

// ---------------- zero workspace ----------------
__global__ __launch_bounds__(256) void zero_kernel(float* p, long long n) {
  long long i = (long long)blockIdx.x * 256 + threadIdx.x;
  if (i < n) p[i] = 0.f;
}

// ---------------- weight convert + transpose: WT[n][k] = (f16)W[k][n] ----------------
__global__ __launch_bounds__(256) void wcvt_kernel(const float* __restrict__ W,
                                                   _Float16* __restrict__ WT,
                                                   int Kd, int Nw) {
  int i = blockIdx.x * 256 + threadIdx.x;
  if (i < Kd * Nw) {
    int n = i / Kd, k = i - n * Kd;
    WT[i] = (_Float16)W[(size_t)k * Nw + n];
  }
}

// ---------------- residual mix + V_in build ----------------
__global__ __launch_bounds__(128) void mix_kernel(
    const float* __restrict__ V, const float* __restrict__ block0,
    const float* __restrict__ s_enc, const float* __restrict__ w_res,
    float* __restrict__ Vin, float* __restrict__ hbuf) {
  __shared__ float red[128];
  int row = blockIdx.x;  // b*N + n
  int c = threadIdx.x;
  float x0 = block0[(size_t)row * D_ + c];
  float x1 = V[(size_t)row * D_ + c];
  float ss0 = block_reduce128(x0 * x0, red, c);
  float ss1 = block_reduce128(x1 * x1, red, c);
  float r0 = rsqrtf(ss0 * (1.f / 128.f) + 1e-8f);
  float r1 = rsqrtf(ss1 * (1.f / 128.f) + 1e-8f);
  float w = w_res[c];
  float d0 = block_reduce128(w * x0 * r0, red, c);
  float d1 = block_reduce128(w * x1 * r1, red, c);
  float mx = fmaxf(d0, d1);
  float e0 = __expf(d0 - mx), e1 = __expf(d1 - mx);
  float a0 = e0 / (e0 + e1);
  float h = a0 * x0 + (1.f - a0) * x1;
  Vin[(size_t)row * NODE_ + c] = h;
  hbuf[(size_t)row * D_ + c] = h;
  if (c < S_) Vin[(size_t)row * NODE_ + D_ + c] = s_enc[(size_t)row * S_ + c];
}

// ---------------- generic layernorm ----------------
__global__ __launch_bounds__(128) void ln_kernel(
    const float* __restrict__ x, const float* __restrict__ g,
    const float* __restrict__ b, float* __restrict__ y) {
  __shared__ float red[128];
  int row = blockIdx.x;
  int c = threadIdx.x;
  float v = x[(size_t)row * D_ + c];
  float mu = block_reduce128(v, red, c) * (1.f / 128.f);
  float dv = v - mu;
  float var = block_reduce128(dv * dv, red, c) * (1.f / 128.f);
  y[(size_t)row * D_ + c] = dv * rsqrtf(var + 1e-5f) * g[c] + b[c];
}

// ---------------- generic WMMA GEMM: Y = act(X@W + bias) (+R), B from global f16 WT ----------------
__global__ __launch_bounds__(256) void gemm128_kernel(
    const float* __restrict__ X, long long xbs, int ldx,
    const _Float16* __restrict__ WT,
    const float* __restrict__ bias,
    float* __restrict__ Y, long long ybs, int ldy,
    const float* __restrict__ R, long long rbs, int ldr,
    int Kdim, int act) {
  __shared__ _Float16 As[2 * 128 * 32] __attribute__((aligned(16)));
  int b = blockIdx.z;
  int row0 = blockIdx.x * 128;
  int nc0 = blockIdx.y * 128;
  int tid = threadIdx.x, wave = tid >> 5, lane = tid & 31;
  const float* Xb = X + (size_t)b * xbs + (size_t)row0 * ldx;

  auto stage = [&](int buf, int k0) {
    _Float16* dst = As + buf * 4096;
    for (int i = tid * 4; i < 128 * 32; i += 1024) {
      int r = i >> 5, k = i & 31;
      const float* src = Xb + (size_t)r * ldx + k0 + k;
      float4 v = *(const float4*)src;
      if (k0 + 96 <= Kdim) __builtin_prefetch(src + 64, 0, 1);
      v4h hv = { (_Float16)v.x, (_Float16)v.y, (_Float16)v.z, (_Float16)v.w };
      *(v4h*)(dst + r * 32 + k) = hv;
    }
  };

  v8f zero = {};
  v8f acc[8];
#pragma unroll
  for (int t = 0; t < 8; ++t) acc[t] = zero;

  int nk = Kdim >> 5;
  stage(0, 0);
  __syncthreads();
  for (int j = 0; j < nk; ++j) {
    int cur = j & 1;
    if (j + 1 < nk) stage(1 - cur, (j + 1) * 32);
    v16h a = load_a_frag(As + cur * 4096, wave * 16, 32);
    int k0 = j * 32;
#pragma unroll
    for (int t = 0; t < 8; ++t)
      acc[t] = wmma32(a, load_b_gl(WT, Kdim, nc0 + t * 16, k0), acc[t]);
    __syncthreads();
  }
  int colb = lane & 15;
  int half = (lane >> 4) & 1;
#pragma unroll
  for (int t = 0; t < 8; ++t) {
#pragma unroll
    for (int r = 0; r < 8; ++r) {
      int m = wave * 16 + 8 * half + r;
      int n = t * 16 + colb;
      float v = acc[t][r] + bias[nc0 + n];
      if (act == 1) v = silu_f(v);
      if (R) v += R[(size_t)b * rbs + (size_t)(row0 + m) * ldr + nc0 + n];
      Y[(size_t)b * ybs + (size_t)(row0 + m) * ldy + nc0 + n] = v;
    }
  }
}

// ---------------- fused edge MLP: gather -> 448x128 -> silu -> 128x128 -> LN -> scatter ----------------
__global__ __launch_bounds__(256) void edge_kernel(
    const float* __restrict__ Vin, const int* __restrict__ edges,
    const float* __restrict__ E,
    const _Float16* __restrict__ w1t, const float* __restrict__ b1,
    const _Float16* __restrict__ w2t, const float* __restrict__ b2,
    const float* __restrict__ g, const float* __restrict__ beta,
    float* __restrict__ Eout, float* __restrict__ sums,
    float* __restrict__ cnt0, float* __restrict__ cnt1) {
  __shared__ _Float16 As[2 * 128 * 32] __attribute__((aligned(16)));
  __shared__ _Float16 Hs[128 * 128] __attribute__((aligned(16)));
  int b = blockIdx.y;
  int m0 = blockIdx.x * 128;
  int tid = threadIdx.x, wave = tid >> 5, lane = tid & 31;
  const int* eb = edges + ((size_t)b * M_ + m0) * 2;
  const float* Eb = E + ((size_t)b * M_ + m0) * D_;
  const float* Vb = Vin + (size_t)b * N_ * NODE_;
  int colb = lane & 15;
  int half = (lane >> 4) & 1;

  auto stage = [&](int buf, int k0) {
    _Float16* dst = As + buf * 4096;
    for (int i = tid * 4; i < 128 * 32; i += 1024) {
      int r = i >> 5, k = i & 31;
      int kk = k0 + k;
      const float* src;
      if (kk < 160)      src = Vb + (size_t)eb[r * 2 + 0] * NODE_ + kk;
      else if (kk < 320) src = Vb + (size_t)eb[r * 2 + 1] * NODE_ + (kk - 160);
      else               src = Eb + (size_t)r * D_ + (kk - 320);
      float4 v = *(const float4*)src;
      v4h hv = { (_Float16)v.x, (_Float16)v.y, (_Float16)v.z, (_Float16)v.w };
      *(v4h*)(dst + r * 32 + k) = hv;
    }
  };

  v8f zero = {};
  v8f acc[8];
#pragma unroll
  for (int t = 0; t < 8; ++t) acc[t] = zero;

  // GEMM1: K = 448 (gathered [snd(160) | rcv(160) | E(128)]), double-buffered A
  stage(0, 0);
  __syncthreads();
  for (int j = 0; j < 14; ++j) {
    int cur = j & 1;
    if (j + 1 < 14) stage(1 - cur, (j + 1) * 32);
    v16h a = load_a_frag(As + cur * 4096, wave * 16, 32);
    int k0 = j * 32;
#pragma unroll
    for (int t = 0; t < 8; ++t)
      acc[t] = wmma32(a, load_b_gl(w1t, 448, t * 16, k0), acc[t]);
    __syncthreads();
  }
  // bias + silu -> Hs (f16)
#pragma unroll
  for (int t = 0; t < 8; ++t) {
#pragma unroll
    for (int r = 0; r < 8; ++r) {
      int m = wave * 16 + 8 * half + r;
      int n = t * 16 + colb;
      Hs[m * 128 + n] = (_Float16)silu_f(acc[t][r] + b1[n]);
    }
  }
  __syncthreads();
#pragma unroll
  for (int t = 0; t < 8; ++t) acc[t] = zero;

  // GEMM2: K = 128, A from LDS Hs, B from global WT -> no inner barriers
#pragma unroll
  for (int k0 = 0; k0 < 128; k0 += 32) {
    v16h a = load_a_frag(Hs + k0, wave * 16, 128);
#pragma unroll
    for (int t = 0; t < 8; ++t)
      acc[t] = wmma32(a, load_b_gl(w2t, 128, t * 16, k0), acc[t]);
  }
  // bias + register-resident layernorm + E_new + scatter-mean accumulation
#pragma unroll
  for (int t = 0; t < 8; ++t)
#pragma unroll
    for (int r = 0; r < 8; ++r) acc[t][r] += b2[t * 16 + colb];

#pragma unroll
  for (int r = 0; r < 8; ++r) {
    float s = 0.f, q = 0.f;
#pragma unroll
    for (int t = 0; t < 8; ++t) { float v = acc[t][r]; s += v; q += v * v; }
#pragma unroll
    for (int mk = 1; mk < 16; mk <<= 1) { s += __shfl_xor(s, mk, 16); q += __shfl_xor(q, mk, 16); }
    float mu = s * (1.f / 128.f);
    float var = q * (1.f / 128.f) - mu * mu;
    float rinv = rsqrtf(var + 1e-5f);
    int mloc = wave * 16 + 8 * half + r;
    int snd = eb[mloc * 2 + 0];
    int rcv = eb[mloc * 2 + 1];
    size_t erow = ((size_t)b * M_ + m0 + mloc) * D_;
#pragma unroll
    for (int t = 0; t < 8; ++t) {
      int n = t * 16 + colb;
      float emb = (acc[t][r] - mu) * rinv * g[n] + beta[n];
      Eout[erow + n] = Eb[(size_t)mloc * D_ + n] + emb;
      if (n < 64) atomicAdd(&sums[((size_t)b * N_ + snd) * D_ + n], emb);
      else        atomicAdd(&sums[((size_t)b * N_ + rcv) * D_ + n], emb);
    }
    if (colb == 0) {
      atomicAdd(&cnt0[(size_t)b * N_ + snd], 1.f);
      atomicAdd(&cnt1[(size_t)b * N_ + rcv], 1.f);
    }
  }
}

// ---------------- fused node MLP: [V_in | seg-mean] -> 288x128 -> silu -> 128x128 -> LN -> h += ----------------
__global__ __launch_bounds__(256) void node_kernel(
    const float* __restrict__ Vin, const float* __restrict__ sums,
    const float* __restrict__ cnt0, const float* __restrict__ cnt1,
    const _Float16* __restrict__ w1t, const float* __restrict__ b1,
    const _Float16* __restrict__ w2t, const float* __restrict__ b2,
    const float* __restrict__ g, const float* __restrict__ beta,
    float* __restrict__ hb) {
  __shared__ _Float16 As[2 * 128 * 32] __attribute__((aligned(16)));
  __shared__ _Float16 Hs[128 * 128] __attribute__((aligned(16)));
  int b = blockIdx.y;
  int n0 = blockIdx.x * 128;
  int tid = threadIdx.x, wave = tid >> 5, lane = tid & 31;
  int colb = lane & 15;
  int half = (lane >> 4) & 1;

  auto stage = [&](int buf, int k0) {
    _Float16* dst = As + buf * 4096;
    for (int i = tid * 4; i < 128 * 32; i += 1024) {
      int r = i >> 5, k = i & 31;
      int kk = k0 + k;
      size_t node = (size_t)b * N_ + n0 + r;
      float4 v;
      if (kk < 160) {
        v = *(const float4*)(Vin + node * NODE_ + kk);
      } else if (kk < 224) {
        float ic = 1.f / fmaxf(cnt0[node], 1.f);
        float4 sv = *(const float4*)(sums + node * D_ + (kk - 160));
        v.x = sv.x * ic; v.y = sv.y * ic; v.z = sv.z * ic; v.w = sv.w * ic;
      } else {
        float ic = 1.f / fmaxf(cnt1[node], 1.f);
        float4 sv = *(const float4*)(sums + node * D_ + 64 + (kk - 224));
        v.x = sv.x * ic; v.y = sv.y * ic; v.z = sv.z * ic; v.w = sv.w * ic;
      }
      v4h hv = { (_Float16)v.x, (_Float16)v.y, (_Float16)v.z, (_Float16)v.w };
      *(v4h*)(dst + r * 32 + k) = hv;
    }
  };

  v8f zero = {};
  v8f acc[8];
#pragma unroll
  for (int t = 0; t < 8; ++t) acc[t] = zero;

  stage(0, 0);
  __syncthreads();
  for (int j = 0; j < 9; ++j) {
    int cur = j & 1;
    if (j + 1 < 9) stage(1 - cur, (j + 1) * 32);
    v16h a = load_a_frag(As + cur * 4096, wave * 16, 32);
    int k0 = j * 32;
#pragma unroll
    for (int t = 0; t < 8; ++t)
      acc[t] = wmma32(a, load_b_gl(w1t, 288, t * 16, k0), acc[t]);
    __syncthreads();
  }
#pragma unroll
  for (int t = 0; t < 8; ++t) {
#pragma unroll
    for (int r = 0; r < 8; ++r) {
      int m = wave * 16 + 8 * half + r;
      int n = t * 16 + colb;
      Hs[m * 128 + n] = (_Float16)silu_f(acc[t][r] + b1[n]);
    }
  }
  __syncthreads();
#pragma unroll
  for (int t = 0; t < 8; ++t) acc[t] = zero;

#pragma unroll
  for (int k0 = 0; k0 < 128; k0 += 32) {
    v16h a = load_a_frag(Hs + k0, wave * 16, 128);
#pragma unroll
    for (int t = 0; t < 8; ++t)
      acc[t] = wmma32(a, load_b_gl(w2t, 128, t * 16, k0), acc[t]);
  }
#pragma unroll
  for (int t = 0; t < 8; ++t)
#pragma unroll
    for (int r = 0; r < 8; ++r) acc[t][r] += b2[t * 16 + colb];

#pragma unroll
  for (int r = 0; r < 8; ++r) {
    float s = 0.f, q = 0.f;
#pragma unroll
    for (int t = 0; t < 8; ++t) { float v = acc[t][r]; s += v; q += v * v; }
#pragma unroll
    for (int mk = 1; mk < 16; mk <<= 1) { s += __shfl_xor(s, mk, 16); q += __shfl_xor(q, mk, 16); }
    float mu = s * (1.f / 128.f);
    float var = q * (1.f / 128.f) - mu * mu;
    float rinv = rsqrtf(var + 1e-5f);
    int mloc = wave * 16 + 8 * half + r;
    size_t hrow = ((size_t)b * N_ + n0 + mloc) * D_;
#pragma unroll
    for (int t = 0; t < 8; ++t) {
      int n = t * 16 + colb;
      float vn = (acc[t][r] - mu) * rinv * g[n] + beta[n];
      hb[hrow + n] += vn;
    }
  }
}

// ---------------- attention: S = scale * Q K^T (f16 out; dh=32 -> 1 WMMA k-step) ----------------
__global__ __launch_bounds__(256) void attn_scores_kernel(
    const float* __restrict__ Q, long long qbs, int Lq,
    const float* __restrict__ K, long long kbs, int Lk,
    _Float16* __restrict__ S, float scale) {
  __shared__ _Float16 Qs[128 * 32] __attribute__((aligned(16)));
  __shared__ _Float16 Ks[128 * 32] __attribute__((aligned(16)));
  int bh = blockIdx.z;
  int b = bh / H_, h = bh % H_;
  int q0 = blockIdx.x * 128, k0 = blockIdx.y * 128;
  int tid = threadIdx.x, wave = tid >> 5, lane = tid & 31;
  int hoff = h * 32;
  const float* Qb = Q + (size_t)b * qbs;
  const float* Kb = K + (size_t)b * kbs;
  for (int i = tid * 4; i < 128 * 32; i += 1024) {
    int r = i >> 5, kd = i & 31;
    float4 qv = *(const float4*)(Qb + (size_t)(q0 + r) * D_ + hoff + kd);
    float4 kv = *(const float4*)(Kb + (size_t)(k0 + r) * D_ + hoff + kd);
    v4h qh = { (_Float16)qv.x, (_Float16)qv.y, (_Float16)qv.z, (_Float16)qv.w };
    v4h kh = { (_Float16)kv.x, (_Float16)kv.y, (_Float16)kv.z, (_Float16)kv.w };
    *(v4h*)(Qs + r * 32 + kd) = qh;
    *(v4h*)(Ks + r * 32 + kd) = kh;
  }
  __syncthreads();
  v16h a = load_a_frag(Qs, wave * 16, 32);
  v8f zero = {};
  v8f acc[8];
#pragma unroll
  for (int t = 0; t < 8; ++t) acc[t] = wmma32(a, load_b_frag(Ks, t * 16, 32), zero);
  int colb = lane & 15;
  int half = (lane >> 4) & 1;
  _Float16* Sb = S + (size_t)bh * Lq * Lk;
#pragma unroll
  for (int t = 0; t < 8; ++t) {
#pragma unroll
    for (int r = 0; r < 8; ++r) {
      int m = wave * 16 + 8 * half + r;
      Sb[(size_t)(q0 + m) * Lk + k0 + t * 16 + colb] = (_Float16)(acc[t][r] * scale);
    }
  }
}

// ---------------- row softmax (f16 in place) ----------------
__global__ __launch_bounds__(256) void softmax_kernel(_Float16* __restrict__ S, int Lk) {
  __shared__ float red[256];
  size_t row = blockIdx.x;
  _Float16* p = S + row * (size_t)Lk;
  int tid = threadIdx.x;
  float mx = -1e30f;
  for (int i = tid; i < Lk; i += 256) mx = fmaxf(mx, (float)p[i]);
  red[tid] = mx; __syncthreads();
  for (int s = 128; s > 0; s >>= 1) { if (tid < s) red[tid] = fmaxf(red[tid], red[tid + s]); __syncthreads(); }
  mx = red[0]; __syncthreads();
  float sum = 0.f;
  for (int i = tid; i < Lk; i += 256) { float e = __expf((float)p[i] - mx); p[i] = (_Float16)e; sum += e; }
  red[tid] = sum; __syncthreads();
  for (int s = 128; s > 0; s >>= 1) { if (tid < s) red[tid] += red[tid + s]; __syncthreads(); }
  float inv = 1.f / red[0];
  __syncthreads();
  for (int i = tid; i < Lk; i += 256) p[i] = (_Float16)((float)p[i] * inv);
}

// ---------------- O[:, h*32:+32] = P @ V_h (P already f16: raw b128 staging) ----------------
__global__ __launch_bounds__(256) void attn_av_kernel(
    const _Float16* __restrict__ P, int Lq, int Lk,
    const float* __restrict__ Vv, long long vbs,
    float* __restrict__ O, long long obs) {
  __shared__ _Float16 Ps[2 * 128 * 32] __attribute__((aligned(16)));
  __shared__ _Float16 Vt[2 * 32 * 32] __attribute__((aligned(16)));
  int bh = blockIdx.y;
  int b = bh / H_, h = bh % H_;
  int hoff = h * 32;
  int q0 = blockIdx.x * 128;
  int tid = threadIdx.x, wave = tid >> 5, lane = tid & 31;
  const _Float16* Pb = P + (size_t)bh * Lq * Lk;
  const float* Vb = Vv + (size_t)b * vbs;

  auto stage = [&](int buf, int k0) {
    _Float16* pd = Ps + buf * 4096;
    for (int c = tid; c < 512; c += 256) {
      int r = c >> 2, seg = (c & 3) * 8;
      *(v8h*)(pd + r * 32 + seg) = *(const v8h*)(Pb + (size_t)(q0 + r) * Lk + k0 + seg);
    }
    _Float16* vd = Vt + buf * 1024;
    for (int i = tid; i < 1024; i += 256) {
      int n = i >> 5, k = i & 31;
      vd[n * 32 + k] = (_Float16)Vb[(size_t)(k0 + k) * D_ + hoff + n];
    }
  };

  v8f zero = {};
  v8f acc[2];
  acc[0] = zero; acc[1] = zero;
  int nk = Lk >> 5;
  stage(0, 0);
  __syncthreads();
  for (int j = 0; j < nk; ++j) {
    int cur = j & 1;
    if (j + 1 < nk) stage(1 - cur, (j + 1) * 32);
    v16h a = load_a_frag(Ps + cur * 4096, wave * 16, 32);
    acc[0] = wmma32(a, load_b_frag(Vt + cur * 1024, 0, 32), acc[0]);
    acc[1] = wmma32(a, load_b_frag(Vt + cur * 1024, 16, 32), acc[1]);
    __syncthreads();
  }
  int colb = lane & 15;
  int half = (lane >> 4) & 1;
#pragma unroll
  for (int t = 0; t < 2; ++t) {
#pragma unroll
    for (int r = 0; r < 8; ++r) {
      int m = wave * 16 + 8 * half + r;
      O[(size_t)b * obs + (size_t)(q0 + m) * D_ + hoff + t * 16 + colb] = acc[t][r];
    }
  }
}

// ================= host orchestration =================
extern "C" void kernel_launch(void* const* d_in, const int* in_sizes, int n_in,
                              void* d_out, int out_size, void* d_ws, size_t ws_size,
                              hipStream_t stream) {
  (void)in_sizes; (void)n_in; (void)out_size; (void)ws_size;
  const float* V      = (const float*)d_in[0];
  const float* E      = (const float*)d_in[1];
  const int*   edges  = (const int*)d_in[2];
  const float* s_enc  = (const float*)d_in[3];
  const float* block0 = (const float*)d_in[4];
  const float* w_res  = (const float*)d_in[5];
  const float* fe_w1  = (const float*)d_in[6];
  const float* fe_b1  = (const float*)d_in[7];
  const float* fe_w2  = (const float*)d_in[8];
  const float* fe_b2  = (const float*)d_in[9];
  const float* fe_g   = (const float*)d_in[10];
  const float* fe_be  = (const float*)d_in[11];
  const float* fn_w1  = (const float*)d_in[12];
  const float* fn_b1  = (const float*)d_in[13];
  const float* fn_w2  = (const float*)d_in[14];
  const float* fn_b2  = (const float*)d_in[15];
  const float* fn_g   = (const float*)d_in[16];
  const float* fn_be  = (const float*)d_in[17];
  const float* ln1_g  = (const float*)d_in[18];
  const float* ln1_b  = (const float*)d_in[19];
  const float* ln2_g  = (const float*)d_in[20];
  const float* ln2_b  = (const float*)d_in[21];
  const float* Qp     = (const float*)d_in[22];
  const float* a_wq   = (const float*)d_in[23];
  const float* a_wk   = (const float*)d_in[24];
  const float* a_wv   = (const float*)d_in[25];
  const float* a_bq   = (const float*)d_in[26];
  const float* a_bk   = (const float*)d_in[27];
  const float* a_bv   = (const float*)d_in[28];
  const float* a_wo   = (const float*)d_in[29];
  const float* a_bo   = (const float*)d_in[30];
  const float* ffn_w1 = (const float*)d_in[31];
  const float* ffn_b1 = (const float*)d_in[32];
  const float* ffn_w2 = (const float*)d_in[33];
  const float* ffn_b2 = (const float*)d_in[34];

  float* ws = (float*)d_ws;
  size_t off = 0;
  float* VIN  = ws + off; off += (size_t)B_ * N_ * NODE_;
  float* HBUF = ws + off; off += (size_t)B_ * N_ * D_;
  float* XBUF = ws + off; off += (size_t)B_ * N_ * D_;
  float* KBUF = ws + off; off += (size_t)B_ * N_ * D_;
  float* VBUF = ws + off; off += (size_t)B_ * N_ * D_;   // KBUF|VBUF contiguous -> FFN hidden
  _Float16* SC = (_Float16*)(ws + off); off += (size_t)B_ * H_ * N_ * NT_ / 2;  // f16 scores
  float* QS   = ws + off; off += (size_t)B_ * NT_ * D_;
  float* KSM  = ws + off; off += (size_t)B_ * NT_ * D_;
  float* VSM  = ws + off; off += (size_t)B_ * NT_ * D_;
  float* OSM  = ws + off; off += (size_t)B_ * NT_ * D_;
  float* W1B  = ws + off; off += (size_t)B_ * NT_ * D_;
  float* W2B  = ws + off; off += (size_t)B_ * NT_ * D_;
  float* SUMS = ws + off; off += (size_t)B_ * N_ * D_;
  float* CNT0 = ws + off; off += (size_t)B_ * N_;
  float* CNT1 = ws + off; off += (size_t)B_ * N_;

  // pre-transposed f16 weights
  _Float16* hw = (_Float16*)(ws + off);
  size_t ho = 0;
  auto halloc = [&](size_t n) { _Float16* p = hw + ho; ho += n; return p; };
  _Float16* FE1T = halloc(448 * 128);
  _Float16* FE2T = halloc(128 * 128);
  _Float16* FN1T = halloc(288 * 128);
  _Float16* FN2T = halloc(128 * 128);
  _Float16* AQT  = halloc(3 * 128 * 128);
  _Float16* AKT  = halloc(3 * 128 * 128);
  _Float16* AVT  = halloc(3 * 128 * 128);
  _Float16* AOT  = halloc(3 * 128 * 128);
  _Float16* FW1T = halloc(256 * 128);
  _Float16* FW2T = halloc(128 * 256);

  float* outH = (float*)d_out;
  float* outE = outH + (size_t)B_ * N_ * D_;

  const long long ND  = (long long)N_ * D_;
  const long long N2D = (long long)N_ * 2 * D_;
  const long long TD  = (long long)NT_ * D_;
  const float scl = 0.17677669529663687f;  // 1/sqrt(32)

  auto wcvt = [&](const float* W, _Float16* WT, int Kd, int Nw) {
    int tot = Kd * Nw;
    wcvt_kernel<<<dim3((tot + 255) / 256), 256, 0, stream>>>(W, WT, Kd, Nw);
  };
  auto gemm = [&](const float* X, long long xbs, int ldx,
                  const _Float16* WT, const float* bias,
                  float* Y, long long ybs, int ldy,
                  const float* R, long long rbs, int ldr,
                  int rows, int Kdim, int ncols, int act) {
    dim3 grid(rows / 128, ncols / 128, B_);
    gemm128_kernel<<<grid, 256, 0, stream>>>(X, xbs, ldx, WT, bias, Y, ybs, ldy,
                                             R, rbs, ldr, Kdim, act);
  };

  // 0) one-time (per launch) weight convert+transpose to f16 (0.78MB, stays cached)
  wcvt(fe_w1, FE1T, 448, 128);
  wcvt(fe_w2, FE2T, 128, 128);
  wcvt(fn_w1, FN1T, 288, 128);
  wcvt(fn_w2, FN2T, 128, 128);
  for (int l = 0; l < 3; ++l) {
    wcvt(a_wq + (size_t)l * 16384, AQT + (size_t)l * 16384, 128, 128);
    wcvt(a_wk + (size_t)l * 16384, AKT + (size_t)l * 16384, 128, 128);
    wcvt(a_wv + (size_t)l * 16384, AVT + (size_t)l * 16384, 128, 128);
    wcvt(a_wo + (size_t)l * 16384, AOT + (size_t)l * 16384, 128, 128);
  }
  wcvt(ffn_w1, FW1T, 128, 256);
  wcvt(ffn_w2, FW2T, 256, 128);

  // 1) residual mix -> V_in, h
  mix_kernel<<<B_ * N_, 128, 0, stream>>>(V, block0, s_enc, w_res, VIN, HBUF);

  // 2) zero scatter accumulators (SUMS|CNT0|CNT1 contiguous)
  {
    long long nz = (long long)B_ * N_ * (D_ + 2);
    zero_kernel<<<(unsigned)((nz + 255) / 256), 256, 0, stream>>>(SUMS, nz);
  }

  // 3) edge MLP (fused gather + scatter), writes E_new
  edge_kernel<<<dim3(M_ / 128, B_), 256, 0, stream>>>(
      VIN, edges, E, FE1T, fe_b1, FE2T, fe_b2, fe_g, fe_be, outE, SUMS, CNT0, CNT1);

  // 4) node MLP, h += v_node
  node_kernel<<<dim3(N_ / 128, B_), 256, 0, stream>>>(
      VIN, SUMS, CNT0, CNT1, FN1T, fn_b1, FN2T, fn_b2, fn_g, fn_be, HBUF);

  // 5) x = LN1(h)
  ln_kernel<<<B_ * N_, 128, 0, stream>>>(HBUF, ln1_g, ln1_b, XBUF);

  // 6) MHA1: (Qp -> x)
  gemm(Qp, 0, D_, AQT, a_bq, QS, TD, D_, nullptr, 0, 0, NT_, D_, D_, 0);
  gemm(XBUF, ND, D_, AKT, a_bk, KBUF, ND, D_, nullptr, 0, 0, N_, D_, D_, 0);
  gemm(XBUF, ND, D_, AVT, a_bv, VBUF, ND, D_, nullptr, 0, 0, N_, D_, D_, 0);
  attn_scores_kernel<<<dim3(1, N_ / 128, B_ * H_), 256, 0, stream>>>(QS, TD, NT_, KBUF, ND, N_, SC, scl);
  softmax_kernel<<<B_ * H_ * NT_, 256, 0, stream>>>(SC, N_);
  attn_av_kernel<<<dim3(1, B_ * H_), 256, 0, stream>>>(SC, NT_, N_, VBUF, ND, OSM, TD);
  gemm(OSM, TD, D_, AOT, a_bo, W1B, TD, D_, nullptr, 0, 0, NT_, D_, D_, 0);

  // 7) MHA2: (W -> W) self
  gemm(W1B, TD, D_, AQT + 16384, a_bq + D_, QS, TD, D_, nullptr, 0, 0, NT_, D_, D_, 0);
  gemm(W1B, TD, D_, AKT + 16384, a_bk + D_, KSM, TD, D_, nullptr, 0, 0, NT_, D_, D_, 0);
  gemm(W1B, TD, D_, AVT + 16384, a_bv + D_, VSM, TD, D_, nullptr, 0, 0, NT_, D_, D_, 0);
  attn_scores_kernel<<<dim3(1, 1, B_ * H_), 256, 0, stream>>>(QS, TD, NT_, KSM, TD, NT_, SC, scl);
  softmax_kernel<<<B_ * H_ * NT_, 256, 0, stream>>>(SC, NT_);
  attn_av_kernel<<<dim3(1, B_ * H_), 256, 0, stream>>>(SC, NT_, NT_, VSM, TD, OSM, TD);
  gemm(OSM, TD, D_, AOT + 16384, a_bo + D_, W2B, TD, D_, nullptr, 0, 0, NT_, D_, D_, 0);

  // 8) MHA3: (x -> W), h += out
  gemm(XBUF, ND, D_, AQT + 32768, a_bq + 2 * D_, KBUF, ND, D_, nullptr, 0, 0, N_, D_, D_, 0);
  gemm(W2B, TD, D_, AKT + 32768, a_bk + 2 * D_, KSM, TD, D_, nullptr, 0, 0, NT_, D_, D_, 0);
  gemm(W2B, TD, D_, AVT + 32768, a_bv + 2 * D_, VSM, TD, D_, nullptr, 0, 0, NT_, D_, D_, 0);
  attn_scores_kernel<<<dim3(N_ / 128, 1, B_ * H_), 256, 0, stream>>>(KBUF, ND, N_, KSM, TD, NT_, SC, scl);
  softmax_kernel<<<B_ * H_ * N_, 256, 0, stream>>>(SC, NT_);
  attn_av_kernel<<<dim3(N_ / 128, B_ * H_), 256, 0, stream>>>(SC, N_, NT_, VSM, TD, VBUF, ND);
  gemm(VBUF, ND, D_, AOT + 32768, a_bo + 2 * D_, HBUF, ND, D_, HBUF, ND, D_, N_, D_, D_, 0);

  // 9) y = LN2(h)
  ln_kernel<<<B_ * N_, 128, 0, stream>>>(HBUF, ln2_g, ln2_b, XBUF);

  // 10) FFN: h_out = h + silu(y@w1+b1)@w2+b2  (hidden reuses KBUF|VBUF as (B,N,256))
  float* FFH = KBUF;
  gemm(XBUF, ND, D_, FW1T, ffn_b1, FFH, N2D, 2 * D_, nullptr, 0, 0, N_, D_, 2 * D_, 1);
  gemm(FFH, N2D, 2 * D_, FW2T, ffn_b2, outH, ND, D_, HBUF, ND, D_, N_, 2 * D_, D_, 0);
}
